// GCN_1005022347291
// MI455X (gfx1250) — compile-verified, matrix operand order checked by reading, not run on hardware
//
#include <hip/hip_runtime.h>
#include <hip/hip_bf16.h>

// GCN forward for MI455X (gfx1250, wave32).
// Pipeline: deg/norm -> GEMM(bf16 WMMA, f32 accum) -> [zero; SpMM atomic scatter] x3 -> final GEMM.
// Workspace layout (needs ~412 MB):
//   buf0: N_NODES*512 f32 (204.8 MB)
//   buf1: N_NODES*512 f32 (204.8 MB)
//   deg_out/deg_in/norm_src/norm_dst: 4 * N_NODES f32 (1.6 MB)

#define NNODES 100000
#define DDIM   512
#define BM 128
#define BN 64
#define BK 64     // K-slab per LDS stage: 8 WMMAs between barriers

typedef __bf16 v16bf __attribute__((ext_vector_type(16)));
typedef __bf16 v4bf  __attribute__((ext_vector_type(4)));
typedef float  v8f   __attribute__((ext_vector_type(8)));

__global__ void zero_kernel(float* __restrict__ p, size_t n) {
  size_t i = (size_t)blockIdx.x * blockDim.x + threadIdx.x;
  if (i < n) p[i] = 0.0f;
}

__global__ void deg_kernel(const int* __restrict__ src, const int* __restrict__ dst,
                           float* __restrict__ deg_out, float* __restrict__ deg_in, int nE) {
  int i = blockIdx.x * blockDim.x + threadIdx.x;
  if (i < nE) {
    atomicAdd(&deg_out[src[i]], 1.0f);
    atomicAdd(&deg_in[dst[i]], 1.0f);
  }
}

__global__ void norm_kernel(const float* __restrict__ deg_out, const float* __restrict__ deg_in,
                            float* __restrict__ norm_src, float* __restrict__ norm_dst, int n) {
  int i = blockIdx.x * blockDim.x + threadIdx.x;
  if (i < n) {
    norm_src[i] = rsqrtf(fmaxf(deg_out[i], 1.0f));
    norm_dst[i] = rsqrtf(fmaxf(deg_in[i], 1.0f));
  }
}

// One wave32 per edge: m[dst] += h[src] * norm_src[src].
// Lane l handles float4 at columns {l*4, l*4+128, l*4+256, l*4+384} -> coalesced 512B bursts.
__global__ void spmm_kernel(const float* __restrict__ h, const float* __restrict__ norm_src,
                            const int* __restrict__ src, const int* __restrict__ dst,
                            float* __restrict__ m, int nE) {
  int wave = (blockIdx.x * blockDim.x + threadIdx.x) >> 5;
  int lane = threadIdx.x & 31;
  if (wave >= nE) return;
  int u = src[wave], v = dst[wave];
  float s = norm_src[u];
  const float4* hs = (const float4*)(h + (size_t)u * DDIM);
  float* mv = m + (size_t)v * DDIM;
#pragma unroll
  for (int i = 0; i < 4; ++i) {
    float4 x = hs[lane + i * 32];
    int base = (lane + i * 32) * 4;
    atomicAdd(&mv[base + 0], x.x * s);
    atomicAdd(&mv[base + 1], x.y * s);
    atomicAdd(&mv[base + 2], x.z * s);
    atomicAdd(&mv[base + 3], x.w * s);
  }
}

// C[M,Nout] = relu?( (A .* scaleA_row) @ W + bias )
// Block tile 128x64 (8 wave32 waves, each owns a 16-row strip x 64 cols).
// Software-pipelined: global loads for slab s+1 issue before the WMMAs of slab s.
// LDS rows padded to 72 bf16 (144B = 36 banks) -> conflict-free b128 fragment loads.
// Fragment layouts per CDNA5 ISA 7.12.2 (16-bit A/B, wave32):
//   lane l holds row/col (l&15); half j holds K = j + 8*(j>=8) + 8*(l>=16) (+32 for 2nd slab half)
__global__ __launch_bounds__(256)
void gemm_bf16_kernel(const float* __restrict__ A, const float* __restrict__ scaleA,
                      const float* __restrict__ W, const float* __restrict__ bias,
                      float* __restrict__ C, int M, int Nout, int ldC, int relu) {
  constexpr int LDK = BK + 8;               // padded LDS stride in bf16
  __shared__ __bf16 Asub[BM][LDK];          // 128*72*2 = 18 KB
  __shared__ __bf16 Bsub[BN][LDK];          //  64*72*2 =  9 KB

  const int tid   = threadIdx.x;
  const int wave  = tid >> 5;
  const int lane  = tid & 31;
  const int l15   = lane & 15;
  const int hi16  = lane >> 4;
  const int kb    = hi16 * 8;
  const int mbase = blockIdx.x * BM;
  const int nbase = blockIdx.y * BN;

  // ---- A staging geometry (invariant across K): 8 rows x one float4 each ----
  const int aquad = tid & 15;               // float4 index along the 64-wide K slab
  const int arow0 = tid >> 4;               // 0..15
  bool  aok[8]; float scv[8]; const float4* aptr[8];
#pragma unroll
  for (int p = 0; p < 8; ++p) {
    int grow = mbase + arow0 + p * 16;
    aok[p] = grow < M;
    scv[p] = (aok[p] && scaleA) ? scaleA[grow] : 1.0f;
    aptr[p] = (const float4*)(A + (size_t)(aok[p] ? grow : 0) * DDIM) + aquad;
  }
  // ---- W staging geometry: thread covers col gn; quads {wq, wq+4, wq+8, wq+12},
  //      4 consecutive k per quad -> bf16 results pack into one ds_store_b64 ----
  const int wn = tid & 63;
  const int wq = tid >> 6;                  // 0..3
  const int gn = nbase + wn;
  const bool wok = gn < Nout;

  float4 aReg[8];
  float  wReg[16];

  auto loadTiles = [&](int k0) {
    int q = k0 >> 2;
#pragma unroll
    for (int p = 0; p < 8; ++p)
      aReg[p] = aok[p] ? aptr[p][q] : make_float4(0.f, 0.f, 0.f, 0.f);
#pragma unroll
    for (int p = 0; p < 4; ++p) {
      int kq = (wq + p * 4) * 4;            // quad base k within slab
#pragma unroll
      for (int j = 0; j < 4; ++j)
        wReg[p * 4 + j] = wok ? W[(size_t)(k0 + kq + j) * Nout + gn] : 0.0f;
    }
  };
  auto storeTiles = [&]() {
#pragma unroll
    for (int p = 0; p < 8; ++p) {
      v4bf y;
      y[0] = (__bf16)(aReg[p].x * scv[p]);
      y[1] = (__bf16)(aReg[p].y * scv[p]);
      y[2] = (__bf16)(aReg[p].z * scv[p]);
      y[3] = (__bf16)(aReg[p].w * scv[p]);
      *(v4bf*)&Asub[arow0 + p * 16][aquad * 4] = y;   // 8B aligned (stride 144B)
    }
#pragma unroll
    for (int p = 0; p < 4; ++p) {
      v4bf y;
#pragma unroll
      for (int j = 0; j < 4; ++j) y[j] = (__bf16)wReg[p * 4 + j];
      *(v4bf*)&Bsub[wn][(wq + p * 4) * 4] = y;        // 8B aligned
    }
  };

  v8f acc[4] = {};

  loadTiles(0);
  storeTiles();
  __syncthreads();

  const __bf16* arp = &Asub[wave * 16 + l15][0];
  constexpr int NSTEP = DDIM / BK;          // 8
  for (int s = 0; s < NSTEP; ++s) {
    const bool more = (s + 1) < NSTEP;
    if (more) loadTiles((s + 1) * BK);      // global loads overlap WMMAs below

    v16bf aF[2];
#pragma unroll
    for (int kk = 0; kk < 2; ++kk) {        // two 16x32 A fragments per 64-K slab
      union { v16bf v; __bf16 e[16]; } ua;
#pragma unroll
      for (int j = 0; j < 8; ++j) {
        ua.e[j]     = arp[kk * 32 + kb + j];
        ua.e[8 + j] = arp[kk * 32 + kb + 16 + j];
      }
      aF[kk] = ua.v;
    }
#pragma unroll
    for (int kk = 0; kk < 2; ++kk) {
#pragma unroll
      for (int i = 0; i < 4; ++i) {
        union { v16bf v; __bf16 e[16]; } ub;
        const __bf16* brp = &Bsub[i * 16 + l15][0];
#pragma unroll
        for (int j = 0; j < 8; ++j) {
          ub.e[j]     = brp[kk * 32 + kb + j];
          ub.e[8 + j] = brp[kk * 32 + kb + 16 + j];
        }
        acc[i] = __builtin_amdgcn_wmma_f32_16x16x32_bf16(
            /*neg_a=*/false, aF[kk], /*neg_b=*/false, ub.v,
            /*c_mod=*/(short)0, acc[i], /*reuse_a=*/false, /*reuse_b=*/false);
      }
    }
    __syncthreads();                        // all waves done reading this slab
    if (more) { storeTiles(); __syncthreads(); }
  }

  // ---- epilogue: C/D layout 7.12.2 -> row = wave*16 + hi16*8 + r, col = i*16 + l15 ----
#pragma unroll
  for (int i = 0; i < 4; ++i) {
    int col = nbase + i * 16 + l15;
    if (col < Nout) {
      float bv = bias[col];
#pragma unroll
      for (int r = 0; r < 8; ++r) {
        int row = mbase + wave * 16 + hi16 * 8 + r;
        if (row < M) {
          float val = acc[i][r] + bv;
          if (relu) val = fmaxf(val, 0.0f);
          C[(size_t)row * ldC + col] = val;
        }
      }
    }
  }
}

static inline unsigned cdivu(long long a, long long b) { return (unsigned)((a + b - 1) / b); }

extern "C" void kernel_launch(void* const* d_in, const int* in_sizes, int n_in,
                              void* d_out, int out_size, void* d_ws, size_t ws_size,
                              hipStream_t stream) {
  (void)n_in; (void)out_size; (void)ws_size;
  const float* features = (const float*)d_in[0];
  const int*   src      = (const int*)d_in[1];
  const int*   dst      = (const int*)d_in[2];
  const float* W_lin    = (const float*)d_in[3];
  const float* b_lin    = (const float*)d_in[4];
  const float* W0       = (const float*)d_in[5];
  const float* b0       = (const float*)d_in[6];
  const float* W1       = (const float*)d_in[7];
  const float* b1       = (const float*)d_in[8];
  const float* W2       = (const float*)d_in[9];
  const float* b2       = (const float*)d_in[10];
  float* out = (float*)d_out;

  const int nNodes = NNODES;
  const int nE     = in_sizes[1];
  const size_t hElems = (size_t)nNodes * DDIM;

  char* ws = (char*)d_ws;
  float* buf0 = (float*)ws; ws += hElems * sizeof(float);
  float* buf1 = (float*)ws; ws += hElems * sizeof(float);
  float* deg  = (float*)ws; ws += 2 * (size_t)nNodes * sizeof(float);
  float* norm_src = (float*)ws; ws += (size_t)nNodes * sizeof(float);
  float* norm_dst = (float*)ws;
  float* deg_out_p = deg;
  float* deg_in_p  = deg + nNodes;

  // degrees + norms
  zero_kernel<<<cdivu(2LL * nNodes, 256), 256, 0, stream>>>(deg, (size_t)(2 * nNodes));
  deg_kernel<<<cdivu(nE, 256), 256, 0, stream>>>(src, dst, deg_out_p, deg_in_p, nE);
  norm_kernel<<<cdivu(nNodes, 256), 256, 0, stream>>>(deg_out_p, deg_in_p, norm_src, norm_dst, nNodes);

  auto gemm = [&](const float* Ain, const float* sA, const float* W, const float* b,
                  float* Cout, int Nout, int relu) {
    dim3 grid(cdivu(nNodes, BM), cdivu(Nout, BN));
    gemm_bf16_kernel<<<grid, 256, 0, stream>>>(Ain, sA, W, b, Cout, nNodes, Nout, Nout, relu);
  };
  auto spmm = [&](const float* h, float* m) {
    zero_kernel<<<cdivu((long long)hElems, 256), 256, 0, stream>>>(m, hElems);
    spmm_kernel<<<cdivu((long long)nE * 32, 256), 256, 0, stream>>>(h, norm_src, src, dst, m, nE);
  };

  // h = X @ W_lin + b_lin
  gemm(features, nullptr, W_lin, b_lin, buf0, DDIM, 0);
  // layer 0: relu((norm_dst * SpMM(h*norm_src)) @ W0 + b0)
  spmm(buf0, buf1);
  gemm(buf1, norm_dst, W0, b0, buf0, DDIM, 1);
  // layer 1
  spmm(buf0, buf1);
  gemm(buf1, norm_dst, W1, b1, buf0, DDIM, 1);
  // layer 2 (no relu, 40 classes)
  spmm(buf0, buf1);
  gemm(buf1, norm_dst, W2, b2, out, 40, 0);
}
